// MDCDGCN_63024350101857
// MI455X (gfx1250) — matrix operand order repaired and checked by reading, not compiled, not run on hardware
//
#include <hip/hip_runtime.h>
#include <hip/hip_bf16.h>
#include <math.h>

#define CDIV(a,b) (((a)+(b)-1)/(b))

typedef __attribute__((ext_vector_type(16))) _Float16 v16h;
typedef __attribute__((ext_vector_type(8)))  float    v8f;

// ----------------------------------------------------------------------------
// helpers
// ----------------------------------------------------------------------------
static __device__ __forceinline__ float mishf(float v){
  float sp = (v > 20.f) ? v : log1pf(__expf(v));
  return v * tanhf(sp);
}

static __device__ __forceinline__ unsigned rotl32(unsigned x, int r){
  return (x << r) | (x >> (32 - r));
}

// Threefry-2x32, 20 rounds (JAX-compatible)
static __device__ __forceinline__ void tf2x32(unsigned k0, unsigned k1,
                                              unsigned& x0, unsigned& x1){
  unsigned ks2 = k0 ^ k1 ^ 0x1BD11BDAu;
  x0 += k0; x1 += k1;
  const int RA[4] = {13,15,26,6};
  const int RB[4] = {17,29,16,24};
#pragma unroll
  for (int g = 0; g < 5; ++g){
#pragma unroll
    for (int r = 0; r < 4; ++r){
      int rot = (g & 1) ? RB[r] : RA[r];
      x0 += x1; x1 = rotl32(x1, rot); x1 ^= x0;
    }
    if      (g == 0){ x0 += k1;  x1 += ks2 + 1u; }
    else if (g == 1){ x0 += ks2; x1 += k0  + 2u; }
    else if (g == 2){ x0 += k0;  x1 += k1  + 3u; }
    else if (g == 3){ x0 += k1;  x1 += ks2 + 4u; }
    else            { x0 += ks2; x1 += k0  + 5u; }
  }
}

// WMMA fragment index maps (CDNA5 CDNA5 ISA layouts, wave32)
static __device__ __forceinline__ int a_kidx(int lane, int e){
  return e + ((e < 8) ? 0 : 8) + ((lane < 16) ? 0 : 8);
}
static __device__ __forceinline__ int b_kidx(int lane, int e){
  return e + ((e < 8) ? 0 : 8) + ((lane < 16) ? 0 : 8);
}

// A fragment = two contiguous 8-float runs: [kk+aoff, +8) and [kk+aoff+16, +8)
// (aoff = 0 for lanes 0-15, 8 for lanes 16-31); all 16B-aligned -> 4x b128 loads
static __device__ __forceinline__ v16h load_a_frag_f4(const float* __restrict__ row,
                                                      int kk, int aoff){
  const float4* p0 = (const float4*)(row + kk + aoff);
  const float4* p1 = (const float4*)(row + kk + aoff + 16);
  float4 u0 = p0[0], u1 = p0[1], w0 = p1[0], w1 = p1[1];
  v16h a;
  a[0] =(_Float16)u0.x; a[1] =(_Float16)u0.y; a[2] =(_Float16)u0.z; a[3] =(_Float16)u0.w;
  a[4] =(_Float16)u1.x; a[5] =(_Float16)u1.y; a[6] =(_Float16)u1.z; a[7] =(_Float16)u1.w;
  a[8] =(_Float16)w0.x; a[9] =(_Float16)w0.y; a[10]=(_Float16)w0.z; a[11]=(_Float16)w0.w;
  a[12]=(_Float16)w1.x; a[13]=(_Float16)w1.y; a[14]=(_Float16)w1.z; a[15]=(_Float16)w1.w;
  return a;
}

// ----------------------------------------------------------------------------
// elementwise / reduction kernels
// ----------------------------------------------------------------------------
__global__ void fill_k(float* p, int n, float v){
  int i = blockIdx.x * blockDim.x + threadIdx.x;
  if (i < n) p[i] = v;
}

__global__ void dinv_k(const float* A, float* dinv){
  __shared__ float red[128];
  int i = blockIdx.x;
  red[threadIdx.x] = fmaxf(A[i*128 + threadIdx.x], 0.f);
  __syncthreads();
  for (int s = 64; s > 0; s >>= 1){
    if (threadIdx.x < s) red[threadIdx.x] += red[threadIdx.x + s];
    __syncthreads();
  }
  if (threadIdx.x == 0) dinv[i] = rsqrtf(red[0] + 1e-10f);
}

__global__ void build_L_k(const float* A, const float* dinv, float* L){
  int idx = blockIdx.x * blockDim.x + threadIdx.x;
  if (idx >= 128*128) return;
  int i = idx >> 7, j = idx & 127;
  L[idx] = dinv[i] * fmaxf(A[idx], 0.f) * dinv[j];
}

// spec1: y[b][o][t] = sum_s w[o][s]*x[b][t][s] + bias[o]
__global__ void spec1_k(const float* x, const float* w, const float* bias,
                        float* y, int Bn){
  int idx = blockIdx.x * blockDim.x + threadIdx.x;
  if (idx >= Bn*128*128) return;
  int t = idx & 127, o = (idx >> 7) & 127, b = idx >> 14;
  const float* xr = x + ((size_t)b*128 + t)*5;
  const float* wr = w + o*5;
  float s = bias[o];
#pragma unroll
  for (int ss = 0; ss < 5; ++ss) s += wr[ss] * xr[ss];
  y[idx] = s;
}

// mean/var over (B,T) per channel for an x of shape (B, C, T)
__global__ void bn_stats_k(const float* x, float* mean, float* var,
                           int Bn, int Cn, int T){
  int c = blockIdx.x;
  int n = Bn * T;
  float s = 0.f, s2 = 0.f;
  for (int b = 0; b < Bn; ++b){
    const float* xr = x + ((size_t)b*Cn + c)*T;
    for (int t = threadIdx.x; t < T; t += blockDim.x){
      float v = xr[t];
      s += v; s2 += v*v;
    }
  }
  __shared__ float rs[256], rq[256];
  rs[threadIdx.x] = s; rq[threadIdx.x] = s2;
  __syncthreads();
  for (int st = 128; st > 0; st >>= 1){
    if (threadIdx.x < st){ rs[threadIdx.x]+=rs[threadIdx.x+st]; rq[threadIdx.x]+=rq[threadIdx.x+st]; }
    __syncthreads();
  }
  if (threadIdx.x == 0){
    float m = rs[0] / (float)n;
    mean[c] = m;
    var[c]  = fmaxf(rq[0] / (float)n - m*m, 0.f);
  }
}

// y = mish(g*(x-m)*rsqrt(v+eps)+be); accum? y+= : y=
__global__ void bn_mish_k(const float* x, float* y, const float* mean,
                          const float* var, const float* g, const float* be,
                          int Bn, int Cn, int T, int accum){
  int idx = blockIdx.x * blockDim.x + threadIdx.x;
  if (idx >= Bn*Cn*T) return;
  int c = (idx / T) % Cn;
  float inv = rsqrtf(var[c] + 1e-5f);
  float v = g[c] * (x[idx] - mean[c]) * inv + be[c];
  v = mishf(v);
  if (accum) y[idx] += v; else y[idx] = v;
}

// dst[b][t][c] = src[b][c][t]
__global__ void transpose_k(const float* src, float* dst, int Bn, int Cn, int T){
  int idx = blockIdx.x * blockDim.x + threadIdx.x;
  if (idx >= Bn*Cn*T) return;
  int b = idx / (Cn*T);
  int r = idx - b*Cn*T;
  int t = r / Cn, c = r - t*Cn;
  dst[idx] = src[((size_t)b*Cn + c)*T + t];
}

// affine: y[b][g][t] = sum_cc aw[g][cc][0]*xgL[b][2g+cc][t] + aw[g][cc][1]*xg[b][2g+cc][t] + ab[g]
__global__ void affine_k(const float* xgL, const float* xg, const float* aw,
                         const float* ab, float* y, int Bn){
  int idx = blockIdx.x * blockDim.x + threadIdx.x;
  if (idx >= Bn*64*64) return;
  int b = idx / 4096, r = idx & 4095, g = r >> 6, t = r & 63;
  float s = ab[g];
#pragma unroll
  for (int cc = 0; cc < 2; ++cc){
    int node = 2*g + cc;
    s += aw[g*4 + cc*2 + 0] * xgL[((size_t)b*128 + node)*64 + t];
    s += aw[g*4 + cc*2 + 1] * xg [((size_t)b*128 + node)*64 + t];
  }
  y[idx] = s;
}

// xf[b][c][o] = 0.5*( xga[b][c][o] + (1/3)*pool_{H->64}(acc[b][c][:])[o] )
__global__ void pool_combine_k(const float* xga, const float* acc, float* xf,
                               int Bn, int H){
  int idx = blockIdx.x * blockDim.x + threadIdx.x;
  if (idx >= Bn*64*64) return;
  int b = idx / 4096, r = idx & 4095, c = r >> 6, o = r & 63;
  int s = (o*H) / 64;
  int e = ((o+1)*H + 63) / 64;
  float sum = 0.f;
  for (int t = s; t < e; ++t) sum += acc[((size_t)b*64 + c)*H + t];
  float xcv = (sum / (float)(e - s)) * (1.0f/3.0f);
  xf[idx] = 0.5f * (xga[idx] + xcv);
}

// ----------------------------------------------------------------------------
// deform-conv sampling-parameter prep (Threefry offsets, W=1 bilinear collapse)
// ----------------------------------------------------------------------------
struct __align__(16) SampP { float w0, w1; int y0, y1; };

__global__ void prep_samp_k(SampP* sp, int Bn, int K, int H, int ksz, int fold){
  int idx = blockIdx.x * blockDim.x + threadIdx.x;
  if (idx >= Bn*K*H) return;
  int h = idx % H;
  int kap = (idx / H) % K;
  int b = idx / (K*H);

  // folded key: threefry(key=(0,1234), x=(0, fold))
  unsigned f0 = 0u, f1 = (unsigned)fold;
  tf2x32(0u, 1234u, f0, f1);

  unsigned n = (unsigned)(Bn * 2 * K * H);
  unsigned half = n >> 1;
  auto gen = [&](unsigned flat) -> float {
    unsigned c0, c1; int word;
    if (flat < half){ c0 = flat;        c1 = flat + half; word = 0; }
    else            { c0 = flat - half; c1 = flat;        word = 1; }
    tf2x32(f0, f1, c0, c1);
    unsigned bits = word ? c1 : c0;
    float f = __uint_as_float((bits >> 9) | 0x3f800000u) - 1.0f;  // [0,1)
    return fmaxf(-1.0f, f * 2.0f - 1.0f);                          // [-1,1)
  };
  unsigned flat_dy = (unsigned)(((b*2*K + 2*kap    )*H) + h);
  unsigned flat_dx = (unsigned)(((b*2*K + 2*kap + 1)*H) + h);
  float dy = gen(flat_dy);
  float dx = gen(flat_dx);

  int pad = ksz / 2;
  int ki = kap / ksz, kj = kap - ki*ksz;
  float py = (float)(h - pad + ki) + dy;
  float px = (float)(  - pad + kj) + dx;
  float y0f = floorf(py), x0f = floorf(px);
  float wy = py - y0f, wx = px - x0f;
  float hw = 0.f;
  if (x0f ==  0.0f) hw += (1.0f - wx);
  if (x0f == -1.0f) hw += wx;
  int y0 = (int)y0f;
  SampP p;
  p.w0 = (y0   >= 0 && y0   < H) ? hw * (1.0f - wy) : 0.0f;
  p.w1 = (y0+1 >= 0 && y0+1 < H) ? hw * wy          : 0.0f;
  p.y0 = min(max(y0,     0), H-1);
  p.y1 = min(max(y0 + 1, 0), H-1);
  sp[idx] = p;
}

// ----------------------------------------------------------------------------
// WMMA kernels
// ----------------------------------------------------------------------------
// generic batched GEMM: D[b](M,N) = A[b](M,K) * B[b](K,N) (+bias), f16 in / f32 acc
__global__ void gemm_wmma_k(const float* __restrict__ A, const float* __restrict__ Bm,
                            const float* __restrict__ bias, float* __restrict__ D,
                            int M, int N, int Kd,
                            long sAb, int ldA, int stA,
                            long sBb, int ldB, int stB,
                            long sDb, int biasMode){
  int b    = blockIdx.x;
  int wave = threadIdx.x >> 5;
  int lane = threadIdx.x & 31;
  int tN = N >> 4, tM = M >> 4;
  int tile = blockIdx.y * 8 + wave;
  if (tile >= tM * tN) return;
  int mt = tile / tN, nt = tile % tN;
  const float* Ab = A  + (size_t)b * sAb;
  const float* Bb = Bm + (size_t)b * sBb;
  int m = mt*16 + (lane & 15);
  int n = nt*16 + (lane & 15);
  v8f c = {};
  for (int kk = 0; kk < Kd; kk += 32){
    v16h af, bf;
#pragma unroll
    for (int e = 0; e < 16; ++e){
      int ka = kk + a_kidx(lane, e);
      af[e] = (_Float16)Ab[(size_t)m * ldA + (size_t)ka * stA];
      int kb = kk + b_kidx(lane, e);
      bf[e] = (_Float16)Bb[(size_t)kb * ldB + (size_t)n * stB];
    }
    c = __builtin_amdgcn_wmma_f32_16x16x32_f16(false, af, false, bf,
                                               (short)0, c, false, false);
  }
#pragma unroll
  for (int v = 0; v < 8; ++v){
    int mg = mt*16 + v + ((lane < 16) ? 0 : 8);
    int ng = nt*16 + (lane & 15);
    float val = c[v];
    if      (biasMode == 1) val += bias[ng];
    else if (biasMode == 2) val += bias[mg];
    D[(size_t)b * sDb + (size_t)mg * N + ng] = val;
  }
}

// fused deformable conv as GEMM: per (b,g): out(32,H) = W(32, Cg*KK) @ samp(Cg*KK, H)
// One wave owns one N-tile; a single on-the-fly B fragment feeds both M-tiles
// (2 WMMAs per K-step). KK is compile-time -> kb/KK is a mul-shift.
template<int KK>
__global__ void deform_wmma_k(const float* __restrict__ X, const float* __restrict__ W,
                              const float* __restrict__ bias, const SampP* __restrict__ sp,
                              float* __restrict__ out, int C, int H){
  int bg = blockIdx.x;
  int b = bg >> 1, g = bg & 1;
  int wave = threadIdx.x >> 5, lane = threadIdx.x & 31;
  int nw = blockDim.x >> 5;
  int Cg = C >> 1;
  int Kd = Cg * KK;                 // multiple of 32 for all shapes here
  int NT = H >> 4;
  const float* Xb  = X  + ((size_t)b * C + (size_t)g * Cg) * H;
  const SampP* spb = sp + (size_t)b * KK * H;
  const float* Wr0 = W + (size_t)(g*32 +      (lane & 15)) * Kd;  // mt=0 row
  const float* Wr1 = W + (size_t)(g*32 + 16 + (lane & 15)) * Kd;  // mt=1 row
  __builtin_prefetch(Wr0, 0, 1);    // -> global_prefetch_b8 (weights stay hot in GL2)
  __builtin_prefetch(Wr1, 0, 1);
  int aoff = (lane < 16) ? 0 : 8;
  int kboff = (lane < 16) ? 0 : 8;
  for (int nt = wave; nt < NT; nt += nw){
    int h = nt*16 + (lane & 15);
    v8f c0 = {}, c1 = {};
    for (int kk = 0; kk < Kd; kk += 32){
      // B fragment (the expensive gather+sample) -- built ONCE per K-step
      v16h bf;
      int kb0 = kk + kboff;
#pragma unroll
      for (int e = 0; e < 16; ++e){
        int kb  = kb0 + e + ((e < 8) ? 0 : 8);
        int cch = kb / KK;            // compile-time divisor
        int kap = kb - cch * KK;
        SampP p = spb[kap * H + h];
        const float* xc = Xb + (size_t)cch * H;
        bf[e] = (_Float16)(p.w0 * xc[p.y0] + p.w1 * xc[p.y1]);
      }
      // A fragments: contiguous weight runs, vectorized b128 loads
      v16h a0 = load_a_frag_f4(Wr0, kk, aoff);
      v16h a1 = load_a_frag_f4(Wr1, kk, aoff);
      c0 = __builtin_amdgcn_wmma_f32_16x16x32_f16(false, a0, false, bf,
                                                  (short)0, c0, false, false);
      c1 = __builtin_amdgcn_wmma_f32_16x16x32_f16(false, a1, false, bf,
                                                  (short)0, c1, false, false);
    }
#pragma unroll
    for (int v = 0; v < 8; ++v){
      int mrow = v + ((lane < 16) ? 0 : 8);
      int hg   = nt*16 + (lane & 15);
      int og0  = g*32 + mrow;
      int og1  = g*32 + 16 + mrow;
      out[((size_t)b * 64 + og0) * H + hg] = c0[v] + bias[og0];
      out[((size_t)b * 64 + og1) * H + hg] = c1[v] + bias[og1];
    }
  }
}

// ----------------------------------------------------------------------------
// head kernels
// ----------------------------------------------------------------------------
__global__ void head1_k(const float* xf, const float* fw, const float* fb,
                        float* xcv, int Bn){
  int idx = blockIdx.x * blockDim.x + threadIdx.x;
  if (idx >= Bn*64) return;
  int b = idx >> 6, c = idx & 63;
  float s = fb[c];
  for (int t = 0; t < 64; ++t)
    s += xf[((size_t)b*64 + t)*64 + c] * fw[c*64 + t];
  xcv[idx] = s;
}

__global__ void head2_k(const float* L, const float* xf, const float* wfa,
                        const float* fpb, float* z, int Bn){
  int idx = blockIdx.x * blockDim.x + threadIdx.x;
  if (idx >= Bn*127) return;
  int b = idx / 127, t = idx - b*127;
  auto zf = [&](int u) -> float {
    if (u < 64){
      float s = 0.f;
      for (int j = 0; j < 64; ++j)
        s += L[31*128 + j] * xf[((size_t)b*64 + j)*64 + u];
      return s;
    }
    return xf[((size_t)b*64 + 31)*64 + (u - 64)];
  };
  z[idx] = wfa[0]*zf(t) + wfa[1]*zf(t+1) + fpb[0];
}

__global__ void head3_k(const float* xcv, const float* z, const float* ow,
                        const float* ob, float* dout, int Bn){
  __shared__ float xo[64];
  int b = blockIdx.x, tid = threadIdx.x;
  int s = (tid*127) / 64;
  int e = ((tid+1)*127 + 63) / 64;
  float sum = 0.f;
  for (int t = s; t < e; ++t) sum += z[b*127 + t];
  xo[tid] = 0.5f * (xcv[b*64 + tid] + sum / (float)(e - s));
  __syncthreads();
  if (tid < 4){
    float acc = ob[tid];
    for (int h = 0; h < 64; ++h) acc += ow[tid*64 + h] * xo[h];
    dout[b*4 + tid] = acc;
  }
}

// ----------------------------------------------------------------------------
// host
// ----------------------------------------------------------------------------
extern "C" void kernel_launch(void* const* d_in, const int* in_sizes, int n_in,
                              void* d_out, int out_size, void* d_ws, size_t ws_size,
                              hipStream_t stream){
  (void)in_sizes; (void)n_in; (void)out_size; (void)ws_size;
  const int B = 64, T = 256;

  const float* X        = (const float*)d_in[0];
  const float* A        = (const float*)d_in[1];
  const float* spec1_w  = (const float*)d_in[2];
  const float* spec1_b  = (const float*)d_in[3];
  const float* spec1_g  = (const float*)d_in[4];
  const float* spec1_be = (const float*)d_in[5];
  const float* spec2_w  = (const float*)d_in[6];
  const float* spec2_b  = (const float*)d_in[7];
  const float* spec2_g  = (const float*)d_in[8];
  const float* spec2_be = (const float*)d_in[9];
  const float* gcn_w    = (const float*)d_in[10];
  const float* gcn_b    = (const float*)d_in[11];
  const float* aff_w    = (const float*)d_in[12];
  const float* aff_b    = (const float*)d_in[13];
  const float* bng_g    = (const float*)d_in[14];
  const float* bng_be   = (const float*)d_in[15];
  const float* bnc_g    = (const float*)d_in[16];
  const float* bnc_be   = (const float*)d_in[17];
  const float* fpc_w    = (const float*)d_in[18];
  const float* fpc_b    = (const float*)d_in[19];
  const float* bnfpc_g  = (const float*)d_in[20];
  const float* bnfpc_be = (const float*)d_in[21];
  const float* fpaff_w  = (const float*)d_in[22];
  const float* fpaff_b  = (const float*)d_in[23];
  const float* bnfpg_g  = (const float*)d_in[24];
  const float* bnfpg_be = (const float*)d_in[25];
  const float* out_w    = (const float*)d_in[26];
  const float* out_b    = (const float*)d_in[27];

  float* W = (float*)d_ws;
  size_t o = 0;
  auto F = [&](size_t n){ size_t r = o; o += (n + 63) & ~(size_t)63; return r; };
  float* pL    = W + F(128*128);
  float* pDinv = W + F(128);
  float* pMean = W + F(256);
  float* pVar  = W + F(256);
  float* pT1   = W + F((size_t)B*128*128);
  float* pT2   = W + F((size_t)B*128*128);
  float* pX0   = W + F((size_t)B*128*128);
  float* pXg   = W + F((size_t)B*128*64);
  float* pXgL  = W + F((size_t)B*128*64);
  float* pXga  = W + F((size_t)B*64*64);
  float* pAcc  = W + F((size_t)B*64*128);
  float* pV    = W + F((size_t)B*64*128);
  float* pXf   = W + F((size_t)B*64*64);
  float* pXcv  = W + F((size_t)B*64);
  float* pZ    = W + F((size_t)B*127);
  SampP* pSp   = (SampP*)(W + F((size_t)B*49*128*4));

  // Laplacian
  dinv_k   <<<128, 128, 0, stream>>>(A, pDinv);
  build_L_k<<<CDIV(128*128, T), T, 0, stream>>>(A, pDinv, pL);

  // spec1 + BN + mish
  spec1_k<<<CDIV(B*128*128, T), T, 0, stream>>>(X, spec1_w, spec1_b, pT1, B);
  bn_stats_k<<<128, 256, 0, stream>>>(pT1, pMean, pVar, B, 128, 128);
  bn_mish_k<<<CDIV(B*128*128, T), T, 0, stream>>>(pT1, pT1, pMean, pVar,
                                                  spec1_g, spec1_be, B, 128, 128, 0);

  // spec2 GEMM (WMMA): A=W2(128x128), B=t1[b], bias per-m
  gemm_wmma_k<<<dim3(B, 8), 256, 0, stream>>>(spec2_w, pT1, spec2_b, pT2,
      128, 128, 128, 0L, 128, 1, (long)(128*128), 128, 1, (long)(128*128), 2);
  bn_stats_k<<<128, 256, 0, stream>>>(pT2, pMean, pVar, B, 128, 128);
  bn_mish_k<<<CDIV(B*128*128, T), T, 0, stream>>>(pT2, pT2, pMean, pVar,
                                                  spec2_g, spec2_be, B, 128, 128, 0);

  // x0 = transpose(t2)
  transpose_k<<<CDIV(B*128*128, T), T, 0, stream>>>(pT2, pX0, B, 128, 128);

  // gcn GEMM (WMMA): A=x0[b](128x128), B(k,n)=gcn_w[n][k], bias per-n
  gemm_wmma_k<<<dim3(B, 4), 256, 0, stream>>>(pX0, gcn_w, gcn_b, pXg,
      128, 64, 128, (long)(128*128), 128, 1, 0L, 1, 128, (long)(128*64), 1);

  // L-mult GEMM (WMMA): A=L, B=xg[b]
  gemm_wmma_k<<<dim3(B, 4), 256, 0, stream>>>(pL, pXg, nullptr, pXgL,
      128, 64, 128, 0L, 128, 1, (long)(128*64), 64, 1, (long)(128*64), 0);

  // affine + BN + mish -> xg_final (pXga)
  affine_k<<<CDIV(B*64*64, T), T, 0, stream>>>(pXgL, pXg, aff_w, aff_b, pXga, B);
  bn_stats_k<<<64, 256, 0, stream>>>(pXga, pMean, pVar, B, 64, 64);
  bn_mish_k<<<CDIV(B*64*64, T), T, 0, stream>>>(pXga, pXga, pMean, pVar,
                                                bng_g, bng_be, B, 64, 64, 0);

  // MDCM iterations
  const int ksizes[3] = {3, 5, 7};
  for (int i = 0; i < 3; ++i){
    const float* Xin = (i == 0) ? pX0 : pXf;
    int C = (i == 0) ? 128 : 64;
    int H = (i == 0) ? 128 : 64;
    int NT = H / 16;
    int dthreads = 32 * ((NT < 8) ? NT : 8);   // no dead waves when H=64
    fill_k<<<CDIV(B*64*H, T), T, 0, stream>>>(pAcc, B*64*H, 0.f);
    for (int j = 0; j < 3; ++j){
      int k = ksizes[j], K = k*k;
      const float* dcw = (const float*)d_in[28 + 2*(i*3 + j)];
      const float* dcb = (const float*)d_in[29 + 2*(i*3 + j)];
      prep_samp_k<<<CDIV(B*K*H, T), T, 0, stream>>>(pSp, B, K, H, k, i*3 + j);
      if (k == 3)
        deform_wmma_k<9> <<<B*2, dthreads, 0, stream>>>(Xin, dcw, dcb, pSp, pV, C, H);
      else if (k == 5)
        deform_wmma_k<25><<<B*2, dthreads, 0, stream>>>(Xin, dcw, dcb, pSp, pV, C, H);
      else
        deform_wmma_k<49><<<B*2, dthreads, 0, stream>>>(Xin, dcw, dcb, pSp, pV, C, H);
      bn_stats_k<<<64, 256, 0, stream>>>(pV, pMean, pVar, B, 64, H);
      bn_mish_k<<<CDIV(B*64*H, T), T, 0, stream>>>(pV, pAcc, pMean, pVar,
          bnc_g + (i*3 + j)*64, bnc_be + (i*3 + j)*64, B, 64, H, 1);
    }
    pool_combine_k<<<CDIV(B*64*64, T), T, 0, stream>>>(pXga, pAcc, pXf, B, H);
  }

  // head: xc branch
  head1_k<<<CDIV(B*64, T), T, 0, stream>>>(pXf, fpc_w, fpc_b, pXcv, B);
  bn_stats_k<<<64, 256, 0, stream>>>(pXcv, pMean, pVar, B, 64, 1);
  bn_mish_k<<<CDIV(B*64, T), T, 0, stream>>>(pXcv, pXcv, pMean, pVar,
                                             bnfpc_g, bnfpc_be, B, 64, 1, 0);

  // head: graph-pool branch
  head2_k<<<CDIV(B*127, T), T, 0, stream>>>(pL, pXf, fpaff_w, fpaff_b, pZ, B);
  bn_stats_k<<<1, 256, 0, stream>>>(pZ, pMean, pVar, B, 1, 127);
  bn_mish_k<<<CDIV(B*127, T), T, 0, stream>>>(pZ, pZ, pMean, pVar,
                                              bnfpg_g, bnfpg_be, B, 1, 127, 0);

  // combine + classifier
  head3_k<<<B, 64, 0, stream>>>(pXcv, pZ, out_w, out_b, (float*)d_out, B);
}
